// Attention_48653389529387
// MI455X (gfx1250) — matrix operand order repaired and checked by reading, not compiled, not run on hardware
//
#include <hip/hip_runtime.h>

typedef __attribute__((ext_vector_type(16))) __bf16        v16bf;
typedef __attribute__((ext_vector_type(8)))  float         v8f;
typedef __attribute__((ext_vector_type(8)))  unsigned int  u32x8;
typedef __attribute__((ext_vector_type(4)))  unsigned int  u32x4;

// Problem shape (fixed by the reference)
constexpr int B_   = 2;
constexpr int S_   = 2048;
constexpr int D_   = 2048;
constexpr int H_   = 16;
constexpr int KVH_ = 4;
constexpr int HD_  = 128;
constexpr int MROWS = B_ * S_;               // 4096 flattened token rows
constexpr int NQKV  = (H_ + 2 * KVH_) * HD_; // 3072 fused qkv columns

// ---------------------------------------------------------------------------
// fp32 -> bf16 round-to-nearest-even
__device__ __forceinline__ unsigned short f2bf(float f) {
  unsigned int u = __float_as_uint(f);
  unsigned int r = u + 0x7FFFu + ((u >> 16) & 1u);
  return (unsigned short)(r >> 16);
}

__device__ __forceinline__ v16bf as_bf16(u32x8 v) {
  return __builtin_bit_cast(v16bf, v);
}

// Load one 16x32 A-fragment / 32x16 B-fragment worth of bf16 (two b128 each)
__device__ __forceinline__ v16bf load_frag_a(const unsigned short* p) {
  // A layout: K = {base+half*8 .. +7, base+16+half*8 .. +7} (caller adds half*8)
  u32x4 a0 = *(const u32x4*)(p);
  u32x4 a1 = *(const u32x4*)(p + 16);
  u32x8 av = {a0[0], a0[1], a0[2], a0[3], a1[0], a1[1], a1[2], a1[3]};
  return as_bf16(av);
}
__device__ __forceinline__ v16bf load_frag_b(const unsigned short* p) {
  // B layout: K = {base+half*8 .. +7, base+16+half*8 .. +7} (caller adds half*8)
  u32x4 b0 = *(const u32x4*)(p);
  u32x4 b1 = *(const u32x4*)(p + 16);
  u32x8 bv = {b0[0], b0[1], b0[2], b0[3], b1[0], b1[1], b1[2], b1[3]};
  return as_bf16(bv);
}

// ---------------------------------------------------------------------------
// Elementwise fp32 -> bf16 cast
__global__ void cast_bf16_kernel(const float* __restrict__ src,
                                 unsigned short* __restrict__ dst, int n) {
  int i = blockIdx.x * blockDim.x + threadIdx.x;
  if (i < n) dst[i] = f2bf(src[i]);
}

// Cast + transpose: src fp32 [K, N] row-major -> dst bf16 [N, K] row-major
__global__ void cast_transpose_kernel(const float* __restrict__ src,
                                      unsigned short* __restrict__ dst,
                                      int K, int N) {
  int i = blockIdx.x * blockDim.x + threadIdx.x;
  if (i >= K * N) return;
  int k = i % K;
  int n = i / K;
  dst[(size_t)n * K + k] = f2bf(src[(size_t)k * N + n]);
}

// ---------------------------------------------------------------------------
// C[M,N](fp32) = A[M,K](bf16,row-major) * Bt[N,K](bf16,row-major transposed B)
// Register-blocked: one wave computes a 32x64 C region (2x4 WMMA tiles),
// so each K=32 step issues 12 b128 loads for 8 WMMAs (A frags reused 4x,
// B frags reused 2x). M % 32 == 0 and N % 64 == 0 are guaranteed here.
__global__ __launch_bounds__(256)
void gemm_bf16_wmma(const unsigned short* __restrict__ A,
                    const unsigned short* __restrict__ Bt,
                    float* __restrict__ C, int M, int N, int K) {
  int wave = threadIdx.x >> 5;
  int lane = threadIdx.x & 31;
  int tile = blockIdx.x * (blockDim.x >> 5) + wave;
  int ntn  = N >> 6;                       // 64-wide N blocks
  int tm   = tile / ntn;
  int tn   = tile - tm * ntn;
  if (tm >= (M >> 5)) return;              // wave-uniform: EXEC stays all-ones

  int row  = lane & 15;
  int half = lane >> 4;

  const unsigned short* aRow0 = A  + (size_t)(tm * 32 + row) * K + half * 8;
  const unsigned short* aRow1 = aRow0 + (size_t)16 * K;
  const unsigned short* bBase = Bt + (size_t)(tn * 64 + row) * K + half * 8;

  v8f acc[2][4];
#pragma unroll
  for (int mi = 0; mi < 2; ++mi)
#pragma unroll
    for (int ni = 0; ni < 4; ++ni)
      acc[mi][ni] = (v8f){0.f, 0.f, 0.f, 0.f, 0.f, 0.f, 0.f, 0.f};

  for (int k0 = 0; k0 < K; k0 += 32) {
    // prefetch two K-steps ahead (global_prefetch_b8; speculative, OOB-safe)
    __builtin_prefetch(aRow0 + k0 + 64, 0, 1);
    __builtin_prefetch(aRow1 + k0 + 64, 0, 1);
    __builtin_prefetch(bBase + k0 + 64, 0, 1);

    v16bf af[2];
    af[0] = load_frag_a(aRow0 + k0);
    af[1] = load_frag_a(aRow1 + k0);
    v16bf bf[4];
#pragma unroll
    for (int ni = 0; ni < 4; ++ni)
      bf[ni] = load_frag_b(bBase + (size_t)ni * 16 * K + k0);

#pragma unroll
    for (int mi = 0; mi < 2; ++mi)
#pragma unroll
      for (int ni = 0; ni < 4; ++ni)
        acc[mi][ni] = __builtin_amdgcn_wmma_f32_16x16x32_bf16(
            false, af[mi], false, bf[ni], (short)0, acc[mi][ni], false, false);
  }

  // C layout: VGPR r -> row (r + half*8), col = lane%16
#pragma unroll
  for (int mi = 0; mi < 2; ++mi) {
#pragma unroll
    for (int ni = 0; ni < 4; ++ni) {
      float* cPtr = C + (size_t)(tm * 32 + mi * 16 + half * 8) * N
                      + tn * 64 + ni * 16 + row;
#pragma unroll
      for (int r = 0; r < 8; ++r) cPtr[(size_t)r * N] = acc[mi][ni][r];
    }
  }
}

// ---------------------------------------------------------------------------
// RoPE + repack: QKV fp32 [B*S, 3072] -> dst bf16 [b, head, s, d]
__global__ void rope_pack_qk(const float* __restrict__ QKV,
                             const float* __restrict__ cosT,
                             const float* __restrict__ sinT,
                             unsigned short* __restrict__ dst,
                             int nHeads, int colOff) {
  int idx = blockIdx.x * blockDim.x + threadIdx.x;
  if (idx >= B_ * nHeads * S_ * HD_) return;
  int d  = idx % HD_;
  int t  = idx / HD_;
  int s  = t % S_;
  int t2 = t / S_;
  int hh = t2 % nHeads;
  int b  = t2 / nHeads;
  const float* srcRow = QKV + (size_t)(b * S_ + s) * NQKV + colOff + hh * HD_;
  float v   = srcRow[d];
  float rot = (d < HD_ / 2) ? -srcRow[d + HD_ / 2] : srcRow[d - HD_ / 2];
  float val = v * cosT[s * HD_ + d] + rot * sinT[s * HD_ + d];
  dst[idx] = f2bf(val);
}

// V repack with transpose: QKV fp32 -> Vt bf16 [b, kvh, d, s]
__global__ void pack_v_kernel(const float* __restrict__ QKV,
                              unsigned short* __restrict__ Vt) {
  int idx = blockIdx.x * blockDim.x + threadIdx.x;
  if (idx >= B_ * KVH_ * HD_ * S_) return;
  int s   = idx % S_;
  int d   = (idx / S_) % HD_;
  int kvh = (idx / (S_ * HD_)) % KVH_;
  int b   = idx / (S_ * HD_ * KVH_);
  float v = QKV[(size_t)(b * S_ + s) * NQKV + (H_ + KVH_) * HD_ + kvh * HD_ + d];
  Vt[idx] = f2bf(v);
}

// ---------------------------------------------------------------------------
// Flash attention: one wave per 16-query tile. 4 waves / block, private LDS slice.
__global__ __launch_bounds__(128)
void attn_wmma_kernel(const unsigned short* __restrict__ Qb,   // [B,H,S,HD]
                      const unsigned short* __restrict__ Kb,   // [B,KVH,S,HD]
                      const unsigned short* __restrict__ Vt,   // [B,KVH,HD,S]
                      const float* __restrict__ amask,         // [B,S]
                      unsigned short* __restrict__ AOb) {      // [B,S,H*HD] bf16
  __shared__ unsigned short pld[4][16][32];   // per-wave P staging (C -> A relayout)
  int wave = threadIdx.x >> 5;
  int lane = threadIdx.x & 31;
  int id   = blockIdx.x * 4 + wave;           // B*H*(S/16) = 4096 tiles
  int qt   = id & 127;
  int h    = (id >> 7) & 15;
  int b    = id >> 11;
  int kvh  = h >> 2;                          // H/KVH = 4
  int row  = lane & 15;
  int half = lane >> 4;

  // Preload 4 Q A-fragments (covers HD=128 as 4 chunks of K=32)
  const unsigned short* qBase =
      Qb + ((size_t)(b * H_ + h) * S_ + qt * 16 + row) * HD_ + half * 8;
  v16bf qf[4];
#pragma unroll
  for (int dc = 0; dc < 4; ++dc) qf[dc] = load_frag_a(qBase + dc * 32);

  const unsigned short* kBase = Kb + (size_t)(b * KVH_ + kvh) * S_ * HD_;
  const unsigned short* vBase = Vt + (size_t)(b * KVH_ + kvh) * HD_ * S_;

  v8f acc[8];
#pragma unroll
  for (int dt = 0; dt < 8; ++dt) acc[dt] = (v8f){0.f,0.f,0.f,0.f,0.f,0.f,0.f,0.f};
  float mrow[8], lrow[8];
#pragma unroll
  for (int r = 0; r < 8; ++r) { mrow[r] = -1e30f; lrow[r] = 0.f; }

  const float scale = 0.08838834764831845f;   // 1/sqrt(128)
  int qEnd = (qt + 1) * 16;

  for (int kb = 0; kb < qEnd; kb += 32) {
    // ---- scores: two 16-key subtiles, 4 WMMAs each (HD=128) ----
    v8f sc[2];
#pragma unroll
    for (int j = 0; j < 2; ++j) {
      int kbj = kb + j * 16;
      v8f s = {0.f, 0.f, 0.f, 0.f, 0.f, 0.f, 0.f, 0.f};
      const unsigned short* kRow = kBase + (size_t)(kbj + row) * HD_ + half * 8;
#pragma unroll
      for (int dc = 0; dc < 4; ++dc) {
        s = __builtin_amdgcn_wmma_f32_16x16x32_bf16(
            false, qf[dc], false, load_frag_b(kRow + dc * 32), (short)0, s,
            false, false);
      }
      int key = kbj + row;                    // this lane's score column
      float padv = (1.0f - amask[b * S_ + key]) * -1e9f;
#pragma unroll
      for (int r = 0; r < 8; ++r) {
        float v = s[r] * scale + padv;
        int qrow = qt * 16 + r + half * 8;
        if (key > qrow) v = -1e30f;           // causal
        s[r] = v;
      }
      sc[j] = s;
    }

    // ---- online softmax over the 32 keys (row-wise, width-16 butterflies) ----
    float corr[8];
#pragma unroll
    for (int r = 0; r < 8; ++r) {
      float mx = fmaxf(sc[0][r], sc[1][r]);
#pragma unroll
      for (int m = 8; m >= 1; m >>= 1) mx = fmaxf(mx, __shfl_xor(mx, m, 16));
      float mNew = fmaxf(mrow[r], mx);
      float c  = __expf(mrow[r] - mNew);
      float p0 = __expf(sc[0][r] - mNew);
      float p1 = __expf(sc[1][r] - mNew);
      sc[0][r] = p0; sc[1][r] = p1;
      float sum = p0 + p1;
#pragma unroll
      for (int m = 8; m >= 1; m >>= 1) sum += __shfl_xor(sum, m, 16);
      lrow[r] = lrow[r] * c + sum;
      mrow[r] = mNew;
      corr[r] = c;
    }
#pragma unroll
    for (int dt = 0; dt < 8; ++dt)
#pragma unroll
      for (int r = 0; r < 8; ++r) acc[dt][r] *= corr[r];

    // ---- P: C-layout -> A-layout through this wave's LDS slice ----
    // (DS ops from one wave are processed in order; no barrier needed.)
#pragma unroll
    for (int j = 0; j < 2; ++j)
#pragma unroll
      for (int r = 0; r < 8; ++r)
        pld[wave][r + half * 8][j * 16 + row] = f2bf(sc[j][r]);

    unsigned short pe[16] __attribute__((aligned(32)));
#pragma unroll
    for (int i = 0; i < 16; ++i) {
      int kk = (i >> 3) * 16 + (i & 7) + half * 8;
      pe[i] = pld[wave][row][kk];
    }
    v16bf pf;
    __builtin_memcpy(&pf, pe, sizeof(pf));

    // ---- PV: 8 d-tiles of 16 cols, one WMMA each ----
#pragma unroll
    for (int dt = 0; dt < 8; ++dt) {
      const unsigned short* vRow =
          vBase + (size_t)(dt * 16 + row) * S_ + kb + half * 8;
      acc[dt] = __builtin_amdgcn_wmma_f32_16x16x32_bf16(
          false, pf, false, load_frag_b(vRow), (short)0, acc[dt], false, false);
    }
  }

  // ---- normalize and emit bf16 rows for the output projection ----
#pragma unroll
  for (int r = 0; r < 8; ++r) {
    float inv = 1.0f / lrow[r];
    int qrow = qt * 16 + r + half * 8;
    unsigned short* o = AOb + (size_t)(b * S_ + qrow) * (H_ * HD_) + h * HD_;
#pragma unroll
    for (int dt = 0; dt < 8; ++dt)
      o[dt * 16 + row] = f2bf(acc[dt][r] * inv);
  }
}

// ---------------------------------------------------------------------------
extern "C" void kernel_launch(void* const* d_in, const int* in_sizes, int n_in,
                              void* d_out, int out_size, void* d_ws, size_t ws_size,
                              hipStream_t stream) {
  (void)in_sizes; (void)n_in; (void)out_size; (void)ws_size;
  const float* x     = (const float*)d_in[0];
  const float* fcos  = (const float*)d_in[1];
  const float* fsin  = (const float*)d_in[2];
  const float* amask = (const float*)d_in[3];
  const float* Wq    = (const float*)d_in[4];
  const float* Wk    = (const float*)d_in[5];
  const float* Wv    = (const float*)d_in[6];
  const float* Wo    = (const float*)d_in[7];
  float* out = (float*)d_out;

  char* ws = (char*)d_ws;
  size_t off = 0;
  auto take = [&](size_t bytes) -> char* {
    char* p = ws + off;
    off += (bytes + 255) & ~(size_t)255;
    return p;
  };
  unsigned short* xb    = (unsigned short*)take((size_t)MROWS * D_ * 2);   // x bf16
  unsigned short* WtQKV = (unsigned short*)take((size_t)NQKV * D_ * 2);    // [Wq|Wk|Wv]^T
  unsigned short* Wot   = (unsigned short*)take((size_t)D_ * D_ * 2);      // Wo^T
  float*          QKV   = (float*)take((size_t)MROWS * NQKV * 4);          // fused proj out
  unsigned short* Qb    = (unsigned short*)take((size_t)B_ * H_ * S_ * HD_ * 2);
  unsigned short* Kb    = (unsigned short*)take((size_t)B_ * KVH_ * S_ * HD_ * 2);
  unsigned short* Vt    = (unsigned short*)take((size_t)B_ * KVH_ * S_ * HD_ * 2);
  unsigned short* AOb   = (unsigned short*)QKV;  // alias: QKV fully consumed first

  const int T = 256;

  // 1) casts / weight transposes
  cast_bf16_kernel<<<(MROWS * D_) / T, T, 0, stream>>>(x, xb, MROWS * D_);
  cast_transpose_kernel<<<(D_ * H_ * HD_) / T, T, 0, stream>>>(
      Wq, WtQKV, D_, H_ * HD_);
  cast_transpose_kernel<<<(D_ * KVH_ * HD_) / T, T, 0, stream>>>(
      Wk, WtQKV + (size_t)H_ * HD_ * D_, D_, KVH_ * HD_);
  cast_transpose_kernel<<<(D_ * KVH_ * HD_) / T, T, 0, stream>>>(
      Wv, WtQKV + (size_t)(H_ + KVH_) * HD_ * D_, D_, KVH_ * HD_);
  cast_transpose_kernel<<<(D_ * D_) / T, T, 0, stream>>>(Wo, Wot, D_, D_);

  // 2) fused QKV projection: [4096,2048] x [2048,3072], 32x64 per wave
  {
    int waves = (MROWS / 32) * (NQKV / 64);  // 128 * 48 = 6144
    gemm_bf16_wmma<<<waves / 8, 256, 0, stream>>>(xb, WtQKV, QKV, MROWS, NQKV, D_);
  }

  // 3) RoPE + head-major repack (Q, K), transpose-pack V
  rope_pack_qk<<<(B_ * H_ * S_ * HD_) / T, T, 0, stream>>>(
      QKV, fcos, fsin, Qb, H_, 0);
  rope_pack_qk<<<(B_ * KVH_ * S_ * HD_) / T, T, 0, stream>>>(
      QKV, fcos, fsin, Kb, KVH_, H_ * HD_);
  pack_v_kernel<<<(B_ * KVH_ * HD_ * S_) / T, T, 0, stream>>>(QKV, Vt);

  // 4) flash attention: 4096 query tiles, 4 waves/block
  attn_wmma_kernel<<<(B_ * H_ * (S_ / 16)) / 4, 128, 0, stream>>>(
      Qb, Kb, Vt, amask, AOb);

  // 5) output projection: [4096,2048] x [2048,2048] -> fp32 d_out, 32x64 per wave
  {
    int waves = (MROWS / 32) * (D_ / 64);    // 128 * 32 = 4096
    gemm_bf16_wmma<<<waves / 8, 256, 0, stream>>>(AOb, Wot, out, MROWS, D_, D_);
  }
}
